// SVDLinearAttention_29059748724854
// MI455X (gfx1250) — matrix operand order repaired and checked by reading, not compiled, not run on hardware
//
#include <hip/hip_runtime.h>
#include <hip/hip_bf16.h>

// ---------------------------------------------------------------------------
// SVD linear attention for MI455X (gfx1250): all GEMMs via v_wmma_f32_16x16x32_bf16
// SVD replaced by subspace iteration on G = x^T x (2 rounds + MGS ortho).
// GEMM core: 128xNT block tile (NT=64/128), double-buffered LDS, b128 traffic,
// mixed f32/bf16 sources & dests (bf16 wherever the only consumer is WMMA).
// ---------------------------------------------------------------------------

#define D_MODEL 768
#define RANK    64
#define SEQ_N   1024
#define BATCH   32
#define BN      (BATCH * SEQ_N)          // 32768
#define QKV_N   (3 * D_MODEL)            // 2304
#define SCALE_F 0.10206207261596577f     // (768/8)^-0.5
#define SQRTD_F 27.712812921102035f      // sqrt(768)
#define PADE    40                       // LDS row pitch in bf16 elems (80B, 16B-aligned)

typedef __attribute__((ext_vector_type(16))) __bf16 v16bf;
typedef __attribute__((ext_vector_type(8)))  float  v8f;

union FragU  { v16bf v; uint4 q[2]; };
union OctU   { uint4 q; __bf16 h[8]; };

__device__ __forceinline__ float clipf(float v) {
  return fminf(fmaxf(v, -100.0f), 100.0f);
}

// ---------------------------------------------------------------------------
// Generic tiled WMMA GEMM:  C = alpha * op(A) * op(B) [+ bias]
//   op(A): (m,k) -> AT ? A[k*lda+m] : A[m*lda+k]   (TA in {float,__bf16})
//   op(B): (k,n) -> BT ? B[n*ldb+k] : B[k*ldb+n]   (TB in {float,__bf16})
//   C element type TC in {float,__bf16}; bias always f32.
// 256 threads = 8 waves (4 x 2); block tile 128 x NT; each wave 32 x NT/2.
// Requires: K % 32 == 0, N % NT == 0, M % 8 == 0 (M guarded via clamp+store mask).
// ---------------------------------------------------------------------------
template<int NT, bool AT, bool BT, bool CLIP, bool BIAS,
         typename TA, typename TB, typename TC>
__global__ __launch_bounds__(256) void wmma_gemm(
    const TA* __restrict__ A, const TB* __restrict__ B,
    const float* __restrict__ bias, TC* __restrict__ C,
    int M, int N, int K, int lda, int ldb, int ldc,
    long sA, long sB, long sC, float alpha)
{
  static_assert(NT == 64 || NT == 128, "NT must be 64 or 128");
  constexpr bool AF32 = sizeof(TA) == 4;
  constexpr bool BF32 = sizeof(TB) == 4;
  constexpr int FB    = NT / 32;          // B fragments per wave
  constexpr int BQ    = NT / 32;          // f32 B staging quads per thread
  constexpr int BQ8   = NT / 64;          // bf16 B staging octs per thread
  constexpr int NQM   = (NT / 4) - 1;     // f32 !BT staging: n-quad mask
  constexpr int NQSH  = (NT == 128) ? 5 : 4;
  constexpr int NOM   = (NT / 8) - 1;     // bf16 !BT staging: n-oct mask
  constexpr int NOSH  = (NT == 128) ? 4 : 3;

  __shared__ __align__(16) __bf16 ldsA[2][128][PADE];
  __shared__ __align__(16) __bf16 ldsB[2][NT][PADE];

  const TA* Ab = A + (long)blockIdx.z * sA;
  const TB* Bb = B + (long)blockIdx.z * sB;
  TC*       Cb = C + (long)blockIdx.z * sC;

  const int t     = threadIdx.x;
  const int lane  = t & 31;
  const int wid   = t >> 5;
  const int waveM = wid & 3;              // 4 wave-rows of 32
  const int waveN = wid >> 2;             // 2 wave-cols of NT/2
  const int m0    = blockIdx.y * 128;
  const int n0    = blockIdx.x * NT;
  const int half  = lane >> 4;
  const int l16   = lane & 15;

  v8f zero = {};
  v8f acc[2][FB];
#pragma unroll
  for (int i = 0; i < 2; ++i)
#pragma unroll
    for (int j = 0; j < FB; ++j) acc[i][j] = zero;

  const int kTiles = K >> 5;
  float4 a4[4], b4[BQ];
  uint4  a8[2], b8[BQ8];

  // ---- stage: global b128 loads along the memory-contiguous axis ----
  auto loadA = [&](int kt) {
    const int k0 = kt << 5;
    if constexpr (AF32) {
#pragma unroll
      for (int i = 0; i < 4; ++i) {
        int f4 = t + (i << 8);                          // 0..1023 float4 slots
        if (AT) {                                       // contiguous along m
          int mq = (f4 & 31) << 2, c = f4 >> 5;
          int gm = m0 + mq; if (gm > M - 4) gm = M - 4; // clamp: no divergence
          a4[i] = *(const float4*)&Ab[(long)(k0 + c) * lda + gm];
        } else {                                        // contiguous along k
          int r = f4 >> 3, c = (f4 & 7) << 2;
          int gm = m0 + r; if (gm > M - 1) gm = M - 1;
          a4[i] = *(const float4*)&Ab[(long)gm * lda + k0 + c];
        }
      }
    } else {
#pragma unroll
      for (int i = 0; i < 2; ++i) {
        int oc = t + (i << 8);                          // 0..511 bf16-oct slots
        if (AT) {                                       // contiguous along m
          int m8 = (oc & 15) << 3, c = oc >> 4;
          int gm = m0 + m8; if (gm > M - 8) gm = M - 8;
          a8[i] = *(const uint4*)&Ab[(long)(k0 + c) * lda + gm];
        } else {                                        // contiguous along k
          int r = oc >> 2, c = (oc & 3) << 3;
          int gm = m0 + r; if (gm > M - 1) gm = M - 1;
          a8[i] = *(const uint4*)&Ab[(long)gm * lda + k0 + c];
        }
      }
    }
  };
  auto loadB = [&](int kt) {
    const int k0 = kt << 5;
    if constexpr (BF32) {
#pragma unroll
      for (int i = 0; i < BQ; ++i) {
        int f4 = t + (i << 8);
        if (BT) {                                       // contiguous along k
          int n = f4 >> 3, c = (f4 & 7) << 2;
          b4[i] = *(const float4*)&Bb[(long)(n0 + n) * ldb + k0 + c];
        } else {                                        // contiguous along n
          int nq = (f4 & NQM) << 2, c = f4 >> NQSH;
          b4[i] = *(const float4*)&Bb[(long)(k0 + c) * ldb + n0 + nq];
        }
      }
    } else {
#pragma unroll
      for (int i = 0; i < BQ8; ++i) {
        int oc = t + (i << 8);
        if (BT) {
          int n = oc >> 2, c = (oc & 3) << 3;
          b8[i] = *(const uint4*)&Bb[(long)(n0 + n) * ldb + k0 + c];
        } else {
          int n8 = (oc & NOM) << 3, c = oc >> NOSH;
          b8[i] = *(const uint4*)&Bb[(long)(k0 + c) * ldb + n0 + n8];
        }
      }
    }
  };
  auto cvt1 = [&](float v) -> __bf16 {
    if (CLIP) v = clipf(v);
    return (__bf16)v;
  };
  auto fixo = [&](OctU& u) {                            // clip a bf16 oct if needed
    if (CLIP) {
#pragma unroll
      for (int e = 0; e < 8; ++e) u.h[e] = (__bf16)clipf((float)u.h[e]);
    }
  };
  auto storeA = [&](int buf) {
    if constexpr (AF32) {
#pragma unroll
      for (int i = 0; i < 4; ++i) {
        int f4 = t + (i << 8);
        if (AT) {                                       // scatter 4 rows, same k
          int mq = (f4 & 31) << 2, c = f4 >> 5;
          ldsA[buf][mq + 0][c] = cvt1(a4[i].x);
          ldsA[buf][mq + 1][c] = cvt1(a4[i].y);
          ldsA[buf][mq + 2][c] = cvt1(a4[i].z);
          ldsA[buf][mq + 3][c] = cvt1(a4[i].w);
        } else {                                        // 4 contiguous bf16 -> b64
          int r = f4 >> 3, c = (f4 & 7) << 2;
          union { __bf16 h[4]; uint2 d; } tmp;
          tmp.h[0] = cvt1(a4[i].x); tmp.h[1] = cvt1(a4[i].y);
          tmp.h[2] = cvt1(a4[i].z); tmp.h[3] = cvt1(a4[i].w);
          *(uint2*)&ldsA[buf][r][c] = tmp.d;
        }
      }
    } else {
#pragma unroll
      for (int i = 0; i < 2; ++i) {
        int oc = t + (i << 8);
        OctU u; u.q = a8[i]; fixo(u);
        if (AT) {                                       // scatter 8 rows, same k
          int m8 = (oc & 15) << 3, c = oc >> 4;
#pragma unroll
          for (int e = 0; e < 8; ++e) ldsA[buf][m8 + e][c] = u.h[e];
        } else {                                        // contiguous -> b128
          int r = oc >> 2, c = (oc & 3) << 3;
          *(uint4*)&ldsA[buf][r][c] = u.q;
        }
      }
    }
  };
  auto storeB = [&](int buf) {
    if constexpr (BF32) {
#pragma unroll
      for (int i = 0; i < BQ; ++i) {
        int f4 = t + (i << 8);
        if (BT) {
          int n = f4 >> 3, c = (f4 & 7) << 2;
          union { __bf16 h[4]; uint2 d; } tmp;
          tmp.h[0] = cvt1(b4[i].x); tmp.h[1] = cvt1(b4[i].y);
          tmp.h[2] = cvt1(b4[i].z); tmp.h[3] = cvt1(b4[i].w);
          *(uint2*)&ldsB[buf][n][c] = tmp.d;
        } else {                                        // scatter 4 n-rows, same k
          int nq = (f4 & NQM) << 2, c = f4 >> NQSH;
          ldsB[buf][nq + 0][c] = cvt1(b4[i].x);
          ldsB[buf][nq + 1][c] = cvt1(b4[i].y);
          ldsB[buf][nq + 2][c] = cvt1(b4[i].z);
          ldsB[buf][nq + 3][c] = cvt1(b4[i].w);
        }
      }
    } else {
#pragma unroll
      for (int i = 0; i < BQ8; ++i) {
        int oc = t + (i << 8);
        OctU u; u.q = b8[i]; fixo(u);
        if (BT) {
          int n = oc >> 2, c = (oc & 3) << 3;
          *(uint4*)&ldsB[buf][n][c] = u.q;
        } else {
          int n8 = (oc & NOM) << 3, c = oc >> NOSH;
#pragma unroll
          for (int e = 0; e < 8; ++e) ldsB[buf][n8 + e][c] = u.h[e];
        }
      }
    }
  };

  loadA(0); loadB(0);
  for (int kt = 0; kt < kTiles; ++kt) {
    const int buf = kt & 1;
    storeA(buf); storeB(buf);
    if (kt + 1 < kTiles) { loadA(kt + 1); loadB(kt + 1); }  // overlap w/ compute
    if (kt + 2 < kTiles) {                                  // global_prefetch_b8
      const int kp = (kt + 2) << 5;
      __builtin_prefetch(AT ? (const void*)&Ab[(long)kp * lda + m0]
                            : (const void*)&Ab[(long)m0 * lda + kp], 0, 1);
      __builtin_prefetch(BT ? (const void*)&Bb[(long)n0 * ldb + kp]
                            : (const void*)&Bb[(long)kp * ldb + n0], 0, 1);
    }
    __syncthreads();

    // ---- fragments: 2x ds_load_b128 each (16B-aligned via PADE=40) ----
    FragU aF[2], bF[FB];
#pragma unroll
    for (int i = 0; i < 2; ++i) {
      int mrow = waveM * 32 + i * 16 + l16;
      // lane<16: K = {0..7,16..23}; lane>=16: K = {8..15,24..31}
      aF[i].q[0] = *(const uint4*)&ldsA[buf][mrow][half * 8];
      aF[i].q[1] = *(const uint4*)&ldsA[buf][mrow][16 + half * 8];
    }
#pragma unroll
    for (int j = 0; j < FB; ++j) {
      int ncol = waveN * (NT / 2) + j * 16 + l16;
      // lane<16: K = {0..7,16..23}; lane>=16: K = {8..15,24..31}
      bF[j].q[0] = *(const uint4*)&ldsB[buf][ncol][half * 8];
      bF[j].q[1] = *(const uint4*)&ldsB[buf][ncol][16 + half * 8];
    }
#pragma unroll
    for (int i = 0; i < 2; ++i)
#pragma unroll
      for (int j = 0; j < FB; ++j)
        acc[i][j] = __builtin_amdgcn_wmma_f32_16x16x32_bf16(
            false, aF[i].v, false, bF[j].v, (short)0, acc[i][j], false, false);
  }

  // ---- epilogue: C layout VGPR r -> row = r + 8*half, col = lane%16 ----
#pragma unroll
  for (int i = 0; i < 2; ++i)
#pragma unroll
    for (int j = 0; j < FB; ++j) {
      int col  = n0 + waveN * (NT / 2) + j * 16 + l16;
      float bv = BIAS ? bias[col] : 0.0f;
#pragma unroll
      for (int r = 0; r < 8; ++r) {
        int row = m0 + waveM * 32 + i * 16 + half * 8 + r;
        if (row < M) Cb[(long)row * ldc + col] = (TC)(acc[i][j][r] * alpha + bv);
      }
    }
}

// ---------------------------------------------------------------------------
// Elementwise f32 -> bf16 conversion.
// ---------------------------------------------------------------------------
__global__ void f32_to_bf16_kernel(const float* __restrict__ in,
                                   __bf16* __restrict__ outp, long total)
{
  long i = (long)blockIdx.x * 256 + threadIdx.x;
  if (i < total) outp[i] = (__bf16)in[i];
}

// ---------------------------------------------------------------------------
// Pack [Wq;Wk;Wv] -> bf16 Wqkv (2304x768) and [bq;bk;bv] -> f32 bqkv (2304).
// ---------------------------------------------------------------------------
__global__ void concat_w_kernel(const float* __restrict__ Wq, const float* __restrict__ Wk,
                                const float* __restrict__ Wv, __bf16* __restrict__ Wqkv)
{
  long idx = (long)blockIdx.x * 256 + threadIdx.x;
  if (idx >= (long)QKV_N * D_MODEL) return;
  int j = (int)(idx / D_MODEL), kk = (int)(idx % D_MODEL);
  const float* src = j < D_MODEL ? Wq : (j < 2 * D_MODEL ? Wk : Wv);
  Wqkv[idx] = (__bf16)src[(long)(j % D_MODEL) * D_MODEL + kk];
}
__global__ void concat_b_kernel(const float* __restrict__ bq, const float* __restrict__ bk,
                                const float* __restrict__ bv, float* __restrict__ bqkv)
{
  int j = blockIdx.x * 256 + threadIdx.x;
  if (j >= QKV_N) return;
  bqkv[j] = j < D_MODEL ? bq[j] : (j < 2 * D_MODEL ? bk[j - D_MODEL] : bv[j - 2 * D_MODEL]);
}

// ---------------------------------------------------------------------------
// Deterministic init of V (768x64, replicated per batch) via integer hash.
// ---------------------------------------------------------------------------
__global__ void vinit_kernel(float* __restrict__ V, int total)
{
  int idx = blockIdx.x * blockDim.x + threadIdx.x;
  if (idx >= total) return;
  int r = (idx / RANK) % D_MODEL;
  int c = idx % RANK;
  unsigned h = (unsigned)(r * 2654435761u) ^ (unsigned)(c * 40503u + 0x9e3779b9u);
  h ^= h >> 16; h *= 0x45d9f3bu; h ^= h >> 13;
  V[idx] = (float)(h & 0xFFFFu) * (1.0f / 32768.0f) - 1.0f;
}

// ---------------------------------------------------------------------------
// Modified Gram-Schmidt: orthonormalize 64 columns of W (768x64) -> V. (f32)
// ---------------------------------------------------------------------------
__global__ __launch_bounds__(256) void mgs_kernel(const float* __restrict__ Wall,
                                                  float* __restrict__ Vall)
{
  const int b = blockIdx.x;
  const float* W = Wall + (long)b * D_MODEL * RANK;
  float*       V = Vall + (long)b * D_MODEL * RANK;
  __shared__ float red[256];
  const int t = threadIdx.x;

  for (int j = 0; j < RANK; ++j) {
    float w[3];
#pragma unroll
    for (int p = 0; p < 3; ++p) w[p] = W[(long)(t + 256 * p) * RANK + j];
    for (int i = 0; i < j; ++i) {
      float part = 0.0f;
#pragma unroll
      for (int p = 0; p < 3; ++p) part += V[(long)(t + 256 * p) * RANK + i] * w[p];
      red[t] = part; __syncthreads();
      for (int s = 128; s > 0; s >>= 1) { if (t < s) red[t] += red[t + s]; __syncthreads(); }
      float d = red[0]; __syncthreads();
#pragma unroll
      for (int p = 0; p < 3; ++p) w[p] -= d * V[(long)(t + 256 * p) * RANK + i];
    }
    float part = 0.0f;
#pragma unroll
    for (int p = 0; p < 3; ++p) part += w[p] * w[p];
    red[t] = part; __syncthreads();
    for (int s = 128; s > 0; s >>= 1) { if (t < s) red[t] += red[t + s]; __syncthreads(); }
    float inv = rsqrtf(fmaxf(red[0], 1e-24f)); __syncthreads();
#pragma unroll
    for (int p = 0; p < 3; ++p) V[(long)(t + 256 * p) * RANK + j] = w[p] * inv;
    __syncthreads();
  }
}

// ---------------------------------------------------------------------------
// Column-normalize Y (1024x64 per batch) in place -> U_k columns unit-norm.
// ---------------------------------------------------------------------------
__global__ __launch_bounds__(256) void colnorm_kernel(float* __restrict__ Y)
{
  const int c = blockIdx.x, b = blockIdx.y, t = threadIdx.x;
  float* col = Y + (long)b * SEQ_N * RANK + c;
  __shared__ float red[256];
  float s = 0.0f;
#pragma unroll
  for (int p = 0; p < 4; ++p) { float v = col[(long)(t + 256 * p) * RANK]; s += v * v; }
  red[t] = s; __syncthreads();
  for (int st = 128; st > 0; st >>= 1) { if (t < st) red[t] += red[t + st]; __syncthreads(); }
  float inv = rsqrtf(fmaxf(red[0], 1e-24f));
#pragma unroll
  for (int p = 0; p < 4; ++p) col[(long)(t + 256 * p) * RANK] *= inv;
}

// ---------------------------------------------------------------------------
// Row l2-normalize * sqrt(D) in place; rows of 768, arbitrary stride & type
// (used on interleaved bf16 qkv buffer, ld = 2304).
// ---------------------------------------------------------------------------
template<typename T>
__global__ __launch_bounds__(256) void l2norm_rows_kernel(T* __restrict__ X, long ld)
{
  T* row = X + (long)blockIdx.x * ld;
  const int t = threadIdx.x;
  __shared__ float red[256];
  float e[3]; float s = 0.0f;
#pragma unroll
  for (int p = 0; p < 3; ++p) { e[p] = (float)row[t + 256 * p]; s += e[p] * e[p]; }
  red[t] = s; __syncthreads();
  for (int st = 128; st > 0; st >>= 1) { if (t < st) red[t] += red[t + st]; __syncthreads(); }
  float n = sqrtf(red[0]);
  float sc = SQRTD_F / fmaxf(n, 1e-12f);
#pragma unroll
  for (int p = 0; p < 3; ++p) row[t + 256 * p] = (T)(e[p] * sc);
}

// ---------------------------------------------------------------------------
// Softmax over 64 with clip(+/-100) and nan_to_num semantics. In place. (f32)
// ---------------------------------------------------------------------------
__global__ __launch_bounds__(64) void softmax64_kernel(float* __restrict__ S)
{
  float* row = S + (long)blockIdx.x * RANK;
  const int t = threadIdx.x;
  __shared__ float red[64];
  float v = clipf(row[t]);
  red[t] = v; __syncthreads();
  for (int s = 32; s > 0; s >>= 1) { if (t < s) red[t] = fmaxf(red[t], red[t + s]); __syncthreads(); }
  float mx = red[0]; __syncthreads();
  float e = __expf(v - mx);
  red[t] = e; __syncthreads();
  for (int s = 32; s > 0; s >>= 1) { if (t < s) red[t] += red[t + s]; __syncthreads(); }
  float a = e / red[0];
  if (isnan(a) || isinf(a)) a = (a < 0.0f && isinf(a)) ? 0.0f : (1.0f / RANK);
  row[t] = a;
}

// ---------------------------------------------------------------------------
// LayerNorm rows of 768: out = (x - mu) * rsqrt(var + eps) * g + b   (f32)
// ---------------------------------------------------------------------------
__global__ __launch_bounds__(256) void layernorm_kernel(const float* __restrict__ X,
                                                        const float* __restrict__ g,
                                                        const float* __restrict__ bta,
                                                        float* __restrict__ Y)
{
  const float* row = X + (long)blockIdx.x * D_MODEL;
  float*       out = Y + (long)blockIdx.x * D_MODEL;
  const int t = threadIdx.x;
  __shared__ float red[256];
  float e[3]; float s = 0.0f;
#pragma unroll
  for (int p = 0; p < 3; ++p) { e[p] = row[t + 256 * p]; s += e[p]; }
  red[t] = s; __syncthreads();
  for (int st = 128; st > 0; st >>= 1) { if (t < st) red[t] += red[t + st]; __syncthreads(); }
  float mu = red[0] * (1.0f / D_MODEL); __syncthreads();
  float vs = 0.0f;
#pragma unroll
  for (int p = 0; p < 3; ++p) { float d = e[p] - mu; vs += d * d; }
  red[t] = vs; __syncthreads();
  for (int st = 128; st > 0; st >>= 1) { if (t < st) red[t] += red[t + st]; __syncthreads(); }
  float inv = rsqrtf(red[0] * (1.0f / D_MODEL) + 1e-5f);
#pragma unroll
  for (int p = 0; p < 3; ++p) {
    int c = t + 256 * p;
    out[c] = (e[p] - mu) * inv * g[c] + bta[c];
  }
}

// ---------------------------------------------------------------------------
// Host orchestration
// ---------------------------------------------------------------------------
extern "C" void kernel_launch(void* const* d_in, const int* in_sizes, int n_in,
                              void* d_out, int out_size, void* d_ws, size_t ws_size,
                              hipStream_t stream)
{
  (void)in_sizes; (void)n_in; (void)out_size; (void)ws_size;
  const float* x    = (const float*)d_in[0];
  const float* Wq   = (const float*)d_in[1];
  const float* bq   = (const float*)d_in[2];
  const float* Wk   = (const float*)d_in[3];
  const float* bk   = (const float*)d_in[4];
  const float* Wv   = (const float*)d_in[5];
  const float* bv   = (const float*)d_in[6];
  const float* Wo   = (const float*)d_in[7];
  const float* bo   = (const float*)d_in[8];
  const float* ln_g = (const float*)d_in[9];
  const float* ln_b = (const float*)d_in[10];
  float* out = (float*)d_out;

  // workspace bump allocation with lifetime-based aliasing
  char* ws = (char*)d_ws;
  size_t off = 0;
  auto alloc = [&](size_t bytes) -> char* {
    char* p = ws + off;
    off += (bytes + 255) & ~(size_t)255;
    return p;
  };
  // big bf16 region time-shared: G_bf (A-B) -> qkv_bf (C-E) -> attnOut_bf + tmpOut (F-G)
  __bf16* packed = (__bf16*)alloc((size_t)BN * QKV_N * 2);   // 151 MB
  __bf16* x_bf   = (__bf16*)alloc((size_t)BN * D_MODEL * 2); // 50 MB
  __bf16* Wqkv   = (__bf16*)alloc((size_t)QKV_N * D_MODEL * 2);
  float*  bqkv   = (float*) alloc((size_t)QKV_N * 4);
  float*  Vb     = (float*) alloc((size_t)BATCH * D_MODEL * RANK * 4);
  float*  Wb     = (float*) alloc((size_t)BATCH * D_MODEL * RANK * 4);
  float*  Ub     = (float*) alloc((size_t)BATCH * SEQ_N * RANK * 4);
  float*  kproj  = (float*) alloc((size_t)BATCH * RANK * D_MODEL * 4);
  float*  vproj  = (float*) alloc((size_t)BATCH * RANK * D_MODEL * 4);
  float*  sc     = (float*) alloc((size_t)BN * RANK * 4);

  __bf16* G_bf    = packed;                          // dead before qkv written
  __bf16* qv      = packed;                          // q col 0, k col 768, v col 1536
  __bf16* attnOut = packed;                          // qkv dead after scores (48 MB)
  float*  tmpOut  = (float*)(packed + (size_t)BN * D_MODEL); // 96 MB after attnOut

  const long sX   = (long)SEQ_N * D_MODEL;
  const long sG   = (long)D_MODEL * D_MODEL;
  const long sV   = (long)D_MODEL * RANK;
  const long sU   = (long)SEQ_N * RANK;
  const long sKP  = (long)RANK * D_MODEL;
  const long sQKV = (long)SEQ_N * QKV_N;
  dim3 blk(256);

  // --- Stage 0: x -> bf16 once (read by G, qkv, Y) ---
  f32_to_bf16_kernel<<<((long)BN * D_MODEL + 255) / 256, blk, 0, stream>>>(
      x, x_bf, (long)BN * D_MODEL);

  // --- Stage A: G_b = clip(x_b)^T clip(x_b)   (768x768, K=1024, batched) ---
  wmma_gemm<128, true, false, true, false, __bf16, __bf16, __bf16>
      <<<dim3(6, 6, BATCH), blk, 0, stream>>>(
      x_bf, x_bf, nullptr, G_bf, D_MODEL, D_MODEL, SEQ_N, D_MODEL, D_MODEL, D_MODEL,
      sX, sX, sG, 1.0f);

  // --- Stage B: subspace iteration, 2 rounds of V <- orth(G V) ---
  vinit_kernel<<<(BATCH * D_MODEL * RANK + 255) / 256, blk, 0, stream>>>(
      Vb, BATCH * D_MODEL * RANK);
  for (int it = 0; it < 2; ++it) {
    wmma_gemm<64, false, false, false, false, __bf16, float, float>
        <<<dim3(1, 6, BATCH), blk, 0, stream>>>(
        G_bf, Vb, nullptr, Wb, D_MODEL, RANK, D_MODEL, D_MODEL, RANK, RANK,
        sG, sV, sV, 1.0f);
    mgs_kernel<<<BATCH, blk, 0, stream>>>(Wb, Vb);
  }
  // U_b = colnormalize(clip(x_b) V_b)   (1024x64, K=768)
  wmma_gemm<64, false, false, true, false, __bf16, float, float>
      <<<dim3(1, 8, BATCH), blk, 0, stream>>>(
      x_bf, Vb, nullptr, Ub, SEQ_N, RANK, D_MODEL, D_MODEL, RANK, RANK,
      sX, sV, sU, 1.0f);
  colnorm_kernel<<<dim3(RANK, BATCH), blk, 0, stream>>>(Ub);

  // --- Stage C: fused qkv = l2norm_scaled(x [Wq;Wk;Wv]^T + b)  (32768x2304) ---
  concat_w_kernel<<<((long)QKV_N * D_MODEL + 255) / 256, blk, 0, stream>>>(Wq, Wk, Wv, Wqkv);
  concat_b_kernel<<<(QKV_N + 255) / 256, blk, 0, stream>>>(bq, bk, bv, bqkv);
  wmma_gemm<128, false, true, false, true, __bf16, __bf16, __bf16>
      <<<dim3(18, 256, 1), blk, 0, stream>>>(
      x_bf, Wqkv, bqkv, qv, BN, QKV_N, D_MODEL, D_MODEL, D_MODEL, QKV_N, 0, 0, 0, 1.0f);
  l2norm_rows_kernel<__bf16><<<BN, blk, 0, stream>>>(qv,               QKV_N);  // q
  l2norm_rows_kernel<__bf16><<<BN, blk, 0, stream>>>(qv + D_MODEL,     QKV_N);  // k
  l2norm_rows_kernel<__bf16><<<BN, blk, 0, stream>>>(qv + 2 * D_MODEL, QKV_N);  // v

  // --- Stage D: k_proj/v_proj = U^T {k,v}  (64x768, K=1024, batched) ---
  wmma_gemm<128, true, false, false, false, float, __bf16, float>
      <<<dim3(6, 1, BATCH), blk, 0, stream>>>(
      Ub, qv + D_MODEL, nullptr, kproj, RANK, D_MODEL, SEQ_N, RANK, QKV_N, D_MODEL,
      sU, sQKV, sKP, 1.0f);
  wmma_gemm<128, true, false, false, false, float, __bf16, float>
      <<<dim3(6, 1, BATCH), blk, 0, stream>>>(
      Ub, qv + 2 * D_MODEL, nullptr, vproj, RANK, D_MODEL, SEQ_N, RANK, QKV_N, D_MODEL,
      sU, sQKV, sKP, 1.0f);

  // --- Stage E: scores = SCALE * q k_proj^T  (1024x64, K=768), softmax ---
  wmma_gemm<64, false, true, false, false, __bf16, float, float>
      <<<dim3(1, 8, BATCH), blk, 0, stream>>>(
      qv, kproj, nullptr, sc, SEQ_N, RANK, D_MODEL, QKV_N, D_MODEL, RANK,
      sQKV, sKP, sU, SCALE_F);
  softmax64_kernel<<<BN, dim3(64), 0, stream>>>(sc);

  // --- Stage F: attn_out = attn v_proj  (1024x768, K=64, batched) ---
  wmma_gemm<128, false, false, false, false, float, float, __bf16>
      <<<dim3(6, 8, BATCH), blk, 0, stream>>>(
      sc, vproj, nullptr, attnOut, SEQ_N, D_MODEL, RANK, RANK, D_MODEL, D_MODEL,
      sU, sKP, sX, 1.0f);

  // --- Stage G: out = LN(attn_out Wo^T + bo) ---
  wmma_gemm<128, false, true, false, true, __bf16, float, float>
      <<<dim3(6, 256, 1), blk, 0, stream>>>(
      attnOut, Wo, bo, tmpOut, BN, D_MODEL, D_MODEL, D_MODEL, D_MODEL, D_MODEL,
      0, 0, 0, 1.0f);
  layernorm_kernel<<<BN, blk, 0, stream>>>(tmpOut, ln_g, ln_b, out);
}
